// self_LSTM2_90907277787361
// MI455X (gfx1250) — matrix operand order repaired and checked by reading, not compiled, not run on hardware
//
#include <hip/hip_runtime.h>

typedef __attribute__((ext_vector_type(16))) __bf16 v16bf;
typedef __attribute__((ext_vector_type(8)))  float  v8f;

constexpr int kB       = 128;   // batch
constexpr int kT       = 512;   // time steps
constexpr int kH       = 128;   // hidden
constexpr int kC       = 128;   // output classes
constexpr int kG       = 512;   // 4*H gate dim
constexpr int kBt      = 16;    // batch tile per workgroup (WMMA M)
constexpr int kNS      = 33;    // attention window slots (W+1)
constexpr int kK2      = 256;   // concat [c1|h2] K dim for LSTM2
constexpr int kNWG     = kB / kBt;   // 8 workgroups
constexpr int kThreads = 256;        // 8 waves (wave32)

// ---------------- LDS layout (bytes) ----------------
constexpr size_t OFF_BUF   = 0;                                   // bf16 [33][16][128] ring of h2
constexpr size_t OFF_A2    = OFF_BUF   + (size_t)kNS*kBt*kH*2;    // bf16 [16][256]  = [c1 | h2_prev]
constexpr size_t OFF_H1B   = OFF_A2    + (size_t)kBt*kK2*2;       // bf16 [16][128]  h1 (A for gates1)
constexpr size_t OFF_YBF   = OFF_H1B   + (size_t)kBt*kH*2;        // bf16 [16][128]  h2+attn (A for FC)
constexpr size_t OFF_GATES = OFF_YBF   + (size_t)kBt*kH*2;        // f32  [16][512]
constexpr size_t OFF_H2F   = OFF_GATES + (size_t)kBt*kG*4;        // f32  [16][128]
constexpr size_t OFF_C1F   = OFF_H2F   + (size_t)kBt*kH*4;        // f32  [16][128]
constexpr size_t OFF_C2F   = OFF_C1F   + (size_t)kBt*kH*4;        // f32  [16][128]
constexpr size_t OFF_SC    = OFF_C2F   + (size_t)kBt*kH*4;        // f32  [33][16] scores / exp weights
constexpr size_t OFF_HD    = OFF_SC    + (size_t)kNS*kBt*4;       // f32  [16] h2·wt_h
constexpr size_t OFF_RR    = OFF_HD    + 64;                      // f32  [33] row reduce (pad 144)
constexpr size_t OFF_B1    = OFF_RR    + 144;                     // f32  [512] bih1+bhh1
constexpr size_t OFF_B2    = OFF_B1    + (size_t)kG*4;            // f32  [512] bih2+bhh2
constexpr size_t OFF_WI1   = OFF_B2    + (size_t)kG*4;            // f32  [512] Wih1 column
constexpr size_t OFF_WTH   = OFF_WI1   + (size_t)kG*4;            // f32  [128] wt_h
constexpr size_t OFF_WTO   = OFF_WTH   + (size_t)kH*4;            // f32  [128] wt_old
constexpr size_t OFF_FCB   = OFF_WTO   + (size_t)kH*4;            // f32  [128] fcb
constexpr size_t OFF_SCAL  = OFF_FCB   + (size_t)kC*4;            // f32  [8] broadcast scalars
constexpr size_t kSmemBytes= OFF_SCAL  + 32;                      // ~214 KB of 320 KB WGP LDS

// ---------------- device helpers ----------------
static __device__ __forceinline__ float sigf(float v)  { return 1.f / (1.f + __expf(-v)); }
static __device__ __forceinline__ float tanhff(float v){ return 2.f / (1.f + __expf(-2.f * v)) - 1.f; }

// B-fragment: D[m][n] = sum_k A[m][k] * W[n][k]  ->  B[k][n] = W[n][k]
// 16-bit B 32x16: lane n = lane&15; element j -> K = k0 + 16*(lane>=16) + j (contiguous 16 elems)
static __device__ __forceinline__ v16bf load_bfrag(const float* W, int ldk, int nbase, int k0, int lane) {
  const float* p = W + (size_t)(nbase + (lane & 15)) * ldk + k0 + ((lane >> 4) << 4);
  v16bf r;
#pragma unroll
  for (int j = 0; j < 16; ++j) r[j] = (__bf16)p[j];
  return r;
}

// Same but reading from concatenated [W0 ; W1] along K (each 128-wide rows)
static __device__ __forceinline__ v16bf load_bfrag_cat(const float* W0, const float* W1,
                                                       int nbase, int k0, int lane) {
  int kk = k0 + ((lane >> 4) << 4);
  const float* base = (kk < 128) ? W0 : (W1 - 128);
  const float* p = base + (size_t)(nbase + (lane & 15)) * 128 + kk;
  v16bf r;
#pragma unroll
  for (int j = 0; j < 16; ++j) r[j] = (__bf16)p[j];
  return r;
}

// 16-bit A 16x32: lane m = lane&15; element j -> K = k0 + 16*(lane>=16) + j (contiguous 16 elems)
static __device__ __forceinline__ v16bf load_afrag(const __bf16* A, int ldk, int k0, int lane) {
  const __bf16* p = A + (size_t)(lane & 15) * ldk + k0 + ((lane >> 4) << 4);
  v16bf r;
#pragma unroll
  for (int j = 0; j < 8; ++j) r[j] = p[j];
#pragma unroll
  for (int j = 0; j < 8; ++j) r[8 + j] = p[8 + j];
  return r;
}

// C/D 16x16 f32: VGPR r -> M = r + 8*(lane>=16), N = lane&15
static __device__ __forceinline__ void store_cfrag(float* D, int ld, int nbase, v8f acc, int lane) {
  int nn = nbase + (lane & 15);
  int mb = (lane >> 4) << 3;
#pragma unroll
  for (int r = 0; r < 8; ++r) D[(size_t)(mb + r) * ld + nn] = acc[r];
}

static __device__ __forceinline__ void grid_barrier(unsigned* cnt, unsigned* gen,
                                                    unsigned target, unsigned nwg) {
  __syncthreads();
  if (threadIdx.x == 0) {
    __threadfence();
    unsigned prev = atomicAdd(cnt, 1u);
    if (prev == nwg - 1u) {
      atomicExch(cnt, 0u);
      atomicAdd(gen, 1u);
    }
    while (atomicAdd(gen, 0u) < target) { __builtin_amdgcn_s_sleep(2); }
  }
  __syncthreads();
  __threadfence();
}

extern "C" __global__ void __launch_bounds__(kThreads, 1)
lstm_attn_persist(const float* __restrict__ x,
                  const float* __restrict__ wih1, const float* __restrict__ whh1,
                  const float* __restrict__ bih1, const float* __restrict__ bhh1,
                  const float* __restrict__ wih2, const float* __restrict__ whh2,
                  const float* __restrict__ bih2, const float* __restrict__ bhh2,
                  const float* __restrict__ w_t,  const float* __restrict__ fcW,
                  const float* __restrict__ fcb,
                  float* __restrict__ out,
                  unsigned* __restrict__ sync_cnt, unsigned* __restrict__ sync_gen,
                  float* __restrict__ gmax, float* __restrict__ gsum) {
  extern __shared__ char smem[];
  __bf16* bufS   = (__bf16*)(smem + OFF_BUF);
  __bf16* a2S    = (__bf16*)(smem + OFF_A2);
  __bf16* h1bS   = (__bf16*)(smem + OFF_H1B);
  __bf16* ybfS   = (__bf16*)(smem + OFF_YBF);
  float*  gatesS = (float*)(smem + OFF_GATES);
  float*  h2fS   = (float*)(smem + OFF_H2F);
  float*  c1fS   = (float*)(smem + OFF_C1F);
  float*  c2fS   = (float*)(smem + OFF_C2F);
  float*  scS    = (float*)(smem + OFF_SC);
  float*  hdS    = (float*)(smem + OFF_HD);
  float*  rrS    = (float*)(smem + OFF_RR);
  float*  b1S    = (float*)(smem + OFF_B1);
  float*  b2S    = (float*)(smem + OFF_B2);
  float*  wi1S   = (float*)(smem + OFF_WI1);
  float*  wthS   = (float*)(smem + OFF_WTH);
  float*  wtoS   = (float*)(smem + OFF_WTO);
  float*  fcbS   = (float*)(smem + OFF_FCB);
  float*  scalS  = (float*)(smem + OFF_SCAL);

  const int tid  = threadIdx.x;
  const int lane = tid & 31;
  const int wid  = tid >> 5;          // 0..7
  const int wg   = blockIdx.x;        // 0..7
  const int b0   = wg * kBt;
  const int n0   = wid * 64;          // this wave's gate-N chunk

  // ---- persistent register-resident weights (bf16 B-fragments) ----
  v16bf w1f[4][4];                    // Whh1 slice: N 64 x K 128
#pragma unroll
  for (int nt = 0; nt < 4; ++nt)
#pragma unroll
    for (int kt = 0; kt < 4; ++kt)
      w1f[nt][kt] = load_bfrag(whh1, kH, n0 + nt * 16, kt * 32, lane);

  v16bf w2f[4][8];                    // [Wih2;Whh2] slice: N 64 x K 256
#pragma unroll
  for (int nt = 0; nt < 4; ++nt)
#pragma unroll
    for (int kt = 0; kt < 8; ++kt)
      w2f[nt][kt] = load_bfrag_cat(wih2, whh2, n0 + nt * 16, kt * 32, lane);

  v16bf wff[4];                       // fcW slice: N 16 x K 128 (per wave)
#pragma unroll
  for (int kt = 0; kt < 4; ++kt)
    wff[kt] = load_bfrag(fcW, kH, wid * 16, kt * 32, lane);

  // ---- init LDS state ----
  for (int i = tid; i < kNS * kBt * kH; i += kThreads) bufS[i] = (__bf16)0.f;
  for (int i = tid; i < kBt * kK2; i += kThreads) a2S[i] = (__bf16)0.f;
  for (int i = tid; i < kBt * kH; i += kThreads) {
    h1bS[i] = (__bf16)0.f; c1fS[i] = 0.f; c2fS[i] = 0.f; h2fS[i] = 0.f;
  }
  for (int i = tid; i < kG; i += kThreads) {
    b1S[i] = bih1[i] + bhh1[i];
    b2S[i] = bih2[i] + bhh2[i];
    wi1S[i] = wih1[i];
  }
  for (int i = tid; i < kH; i += kThreads) {
    wthS[i] = w_t[i];
    wtoS[i] = w_t[kH + i];
    fcbS[i] = fcb[i];
  }
  __syncthreads();

  unsigned bars = 0;

#pragma unroll 1
  for (int t = 0; t < kT; ++t) {
    const int ring = t % kNS;

    // ---- phase 1: gates1 = h1 @ Whh1^T (WMMA bf16) ----
    {
      v8f acc[4] = {};
#pragma unroll
      for (int kt = 0; kt < 4; ++kt) {
        v16bf a = load_afrag(h1bS, kH, kt * 32, lane);
#pragma unroll
        for (int nt = 0; nt < 4; ++nt)
          acc[nt] = __builtin_amdgcn_wmma_f32_16x16x32_bf16(
              false, a, false, w1f[nt][kt], (short)0, acc[nt], false, false);
      }
#pragma unroll
      for (int nt = 0; nt < 4; ++nt) store_cfrag(gatesS, kG, n0 + nt * 16, acc[nt], lane);
    }
    __syncthreads();

    // ---- phase 2: LSTM1 elementwise (x enters as outer product with Wih1 column) ----
    for (int idx = tid; idx < kBt * kH; idx += kThreads) {
      int b = idx >> 7, h = idx & (kH - 1);
      const float* xp = x + (size_t)(b0 + b) * kT;
      float xb = xp[t];
      __builtin_prefetch(xp + t + 1, 0, 0);
      float gi = gatesS[b * kG + h]            + b1S[h]            + xb * wi1S[h];
      float gf = gatesS[b * kG + kH + h]       + b1S[kH + h]       + xb * wi1S[kH + h];
      float gg = gatesS[b * kG + 2 * kH + h]   + b1S[2 * kH + h]   + xb * wi1S[2 * kH + h];
      float go = gatesS[b * kG + 3 * kH + h]   + b1S[3 * kH + h]   + xb * wi1S[3 * kH + h];
      float cn = sigf(gf) * c1fS[idx] + sigf(gi) * tanhff(gg);
      float hn = sigf(go) * tanhff(cn);
      c1fS[idx] = cn;
      h1bS[idx] = (__bf16)hn;
      a2S[b * kK2 + h] = (__bf16)cn;      // LSTM2 input is c1
    }
    __syncthreads();

    // ---- phase 3: gates2 = [c1|h2_prev] @ [Wih2;Whh2]^T ----
    {
      v8f acc[4] = {};
#pragma unroll
      for (int kt = 0; kt < 8; ++kt) {
        v16bf a = load_afrag(a2S, kK2, kt * 32, lane);
#pragma unroll
        for (int nt = 0; nt < 4; ++nt)
          acc[nt] = __builtin_amdgcn_wmma_f32_16x16x32_bf16(
              false, a, false, w2f[nt][kt], (short)0, acc[nt], false, false);
      }
#pragma unroll
      for (int nt = 0; nt < 4; ++nt) store_cfrag(gatesS, kG, n0 + nt * 16, acc[nt], lane);
    }
    __syncthreads();

    // ---- phase 4: LSTM2 elementwise ----
    for (int idx = tid; idx < kBt * kH; idx += kThreads) {
      int b = idx >> 7, h = idx & (kH - 1);
      float gi = gatesS[b * kG + h]          + b2S[h];
      float gf = gatesS[b * kG + kH + h]     + b2S[kH + h];
      float gg = gatesS[b * kG + 2 * kH + h] + b2S[2 * kH + h];
      float go = gatesS[b * kG + 3 * kH + h] + b2S[3 * kH + h];
      float cn = sigf(gf) * c2fS[idx] + sigf(gi) * tanhff(gg);
      float hn = sigf(go) * tanhff(cn);
      c2fS[idx] = cn;
      h2fS[idx] = hn;
      a2S[b * kK2 + kH + h] = (__bf16)hn;   // becomes h2_prev next step
    }
    __syncthreads();

    // ---- phase 5a: hdot[b] = h2[b]·wt_h ----
    if (tid < kBt) {
      float d = 0.f;
#pragma unroll 8
      for (int h = 0; h < kH; ++h) d += h2fS[tid * kH + h] * wthS[h];
      hdS[tid] = d;
    }
    __syncthreads();

    // ---- phase 5b: masked scores over window (ring buffer, logical s -> (t+s)%33) ----
    {
      int smin = (t >= kNS - 1) ? 0 : (kNS - 1 - t);
      for (int idx = tid; idx < kNS * kBt; idx += kThreads) {
        int s = idx >> 4, b = idx & 15;
        float v = -__builtin_inff();
        if (s >= smin) {
          int ph = ring + s; if (ph >= kNS) ph -= kNS;
          const __bf16* bp = bufS + (size_t)ph * (kBt * kH) + b * kH;
          float d = 0.f;
#pragma unroll 8
          for (int h = 0; h < kH; ++h) d += (float)bp[h] * wtoS[h];
          v = d + hdS[b];
        }
        scS[idx] = v;
      }
    }
    __syncthreads();

    // ---- phase 5c: GLOBAL max across all 8 WGs (softmax is over all 33*128 scores) ----
    for (int s = tid; s < kNS; s += kThreads) {
      float m = -__builtin_inff();
      for (int b = 0; b < kBt; ++b) m = fmaxf(m, scS[s * kBt + b]);
      rrS[s] = m;
    }
    __syncthreads();
    if (tid == 0) {
      float m = -__builtin_inff();
      for (int s = 0; s < kNS; ++s) m = fmaxf(m, rrS[s]);
      atomicExch(&gmax[wg], m);
    }
    grid_barrier(sync_cnt, sync_gen, ++bars, kNWG);
    if (tid == 0) {
      float m = -__builtin_inff();
      for (int i = 0; i < kNWG; ++i) m = fmaxf(m, atomicAdd(&gmax[i], 0.f));
      scalS[0] = m;
    }
    __syncthreads();
    float smax = scalS[0];

    // ---- phase 5d: exp + GLOBAL sum ----
    for (int idx = tid; idx < kNS * kBt; idx += kThreads) {
      float v = scS[idx];
      scS[idx] = (v > -1e30f) ? __expf(v - smax) : 0.f;
    }
    __syncthreads();
    for (int s = tid; s < kNS; s += kThreads) {
      float m = 0.f;
      for (int b = 0; b < kBt; ++b) m += scS[s * kBt + b];
      rrS[s] = m;
    }
    __syncthreads();
    if (tid == 0) {
      float m = 0.f;
      for (int s = 0; s < kNS; ++s) m += rrS[s];
      atomicExch(&gsum[wg], m);
    }
    grid_barrier(sync_cnt, sync_gen, ++bars, kNWG);
    if (tid == 0) {
      float m = 0.f;
      for (int i = 0; i < kNWG; ++i) m += atomicAdd(&gsum[i], 0.f);
      scalS[1] = 1.f / m;
    }
    __syncthreads();
    float sinv = scalS[1];

    // ---- phase 7: attn_c, y = h2 + attn, append h2 to ring ----
    for (int idx = tid; idx < kBt * kH; idx += kThreads) {
      int b = idx >> 7, h = idx & (kH - 1);
      float a = 0.f;
#pragma unroll
      for (int s = 0; s < kNS; ++s) {
        int ph = ring + s; if (ph >= kNS) ph -= kNS;
        a += scS[s * kBt + b] * (float)bufS[(size_t)ph * (kBt * kH) + b * kH + h];
      }
      float h2v = h2fS[idx];
      ybfS[idx] = (__bf16)(h2v + a * sinv);
      bufS[(size_t)ring * (kBt * kH) + b * kH + h] = (__bf16)h2v;  // slot (t%33) = oldest
    }
    __syncthreads();

    // ---- phase 8: fused FC: out[b][t][c] = y @ fcW^T + fcb ----
    {
      v8f oacc = {};
#pragma unroll
      for (int kt = 0; kt < 4; ++kt) {
        v16bf a = load_afrag(ybfS, kH, kt * 32, lane);
        oacc = __builtin_amdgcn_wmma_f32_16x16x32_bf16(
            false, a, false, wff[kt], (short)0, oacc, false, false);
      }
      int cidx = wid * 16 + (lane & 15);
      int mb = (lane >> 4) << 3;
      float bv = fcbS[cidx];
#pragma unroll
      for (int r = 0; r < 8; ++r)
        out[((size_t)(b0 + mb + r) * kT + t) * kC + cidx] = oacc[r] + bv;
    }
    __syncthreads();
  }
}

extern "C" void kernel_launch(void* const* d_in, const int* in_sizes, int n_in,
                              void* d_out, int out_size, void* d_ws, size_t ws_size,
                              hipStream_t stream) {
  const float* x    = (const float*)d_in[0];
  const float* wih1 = (const float*)d_in[1];
  const float* whh1 = (const float*)d_in[2];
  const float* bih1 = (const float*)d_in[3];
  const float* bhh1 = (const float*)d_in[4];
  const float* wih2 = (const float*)d_in[5];
  const float* whh2 = (const float*)d_in[6];
  const float* bih2 = (const float*)d_in[7];
  const float* bhh2 = (const float*)d_in[8];
  const float* w_t  = (const float*)d_in[9];
  const float* fcW  = (const float*)d_in[10];
  const float* fcb  = (const float*)d_in[11];
  float* out = (float*)d_out;

  unsigned* cnt = (unsigned*)d_ws;
  unsigned* gen = cnt + 1;
  float* gmax = (float*)((char*)d_ws + 16);
  float* gsum = (float*)((char*)d_ws + 48);

  // Reset the grid-barrier state every launch (graph-capture safe).
  hipMemsetAsync(d_ws, 0, 80, stream);

  hipFuncSetAttribute(reinterpret_cast<const void*>(lstm_attn_persist),
                      hipFuncAttributeMaxDynamicSharedMemorySize, (int)kSmemBytes);

  lstm_attn_persist<<<dim3(kNWG), dim3(kThreads), kSmemBytes, stream>>>(
      x, wih1, whh1, bih1, bhh1, wih2, whh2, bih2, bhh2, w_t, fcW, fcb,
      out, cnt, gen, gmax, gsum);
}
